// Block_71287867179251
// MI455X (gfx1250) — matrix operand order repaired and checked by reading, not compiled, not run on hardware
//
#include <hip/hip_runtime.h>
#include <hip/hip_bf16.h>
#include <math.h>

typedef __bf16 bf16;
typedef __attribute__((ext_vector_type(16))) __bf16 v16bf;
typedef __attribute__((ext_vector_type(8)))  float  v8f;
typedef __attribute__((ext_vector_type(4)))  unsigned int u32x4;
typedef __attribute__((ext_vector_type(8)))  int i32x8;
typedef __attribute__((ext_vector_type(4)))  int i32x4;
typedef int v4i __attribute__((vector_size(16)));   // matches builtin param type

#define HAS_ASYNC_LDS __has_builtin(__builtin_amdgcn_global_load_async_to_lds_b128)
#define HAS_TDM (__has_builtin(__builtin_amdgcn_tensor_load_to_lds) && \
                 __has_builtin(__builtin_amdgcn_s_wait_tensorcnt))
#define HAS_DPP __has_builtin(__builtin_amdgcn_mov_dpp)

__device__ __forceinline__ v8f wmma_bf16(v16bf a, v16bf b, v8f c) {
  return __builtin_amdgcn_wmma_f32_16x16x32_bf16(false, a, false, b, (short)0, c,
                                                 false, false);
}

__device__ __forceinline__ v8f v8f_zero() {
  v8f z;
#pragma unroll
  for (int i = 0; i < 8; ++i) z[i] = 0.0f;
  return z;
}

// ---- async global->LDS (ASYNCcnt path), sync fallback if builtin absent ----
__device__ __forceinline__ void async_copy16(const bf16* g, bf16* l) {
#if HAS_ASYNC_LDS
  __builtin_amdgcn_global_load_async_to_lds_b128(
      (__attribute__((address_space(1))) v4i*)g,
      (__attribute__((address_space(3))) v4i*)l, 0, 0);
#else
  *(uint4*)l = *(const uint4*)g;
#endif
}

template<int N>
__device__ __forceinline__ void wait_async_le() {
#if HAS_ASYNC_LDS
#if __has_builtin(__builtin_amdgcn_s_wait_asynccnt)
  __builtin_amdgcn_s_wait_asynccnt(N);
#else
  if (N == 0) asm volatile("s_wait_asynccnt 0x0" ::: "memory");
  else        asm volatile("s_wait_asynccnt 0x4" ::: "memory");
#endif
  asm volatile("" ::: "memory");
#endif
}

// ---- 16-lane butterfly reductions in pure VALU via DPP16 ----
#if HAS_DPP
template<int C>
__device__ __forceinline__ float dppf(float x) {
  return __int_as_float(
      __builtin_amdgcn_mov_dpp(__float_as_int(x), C, 0xF, 0xF, true));
}
__device__ __forceinline__ float red_max16(float x) {
  x = fmaxf(x, dppf<0xB1>(x));    // quad_perm [1,0,3,2]  (^1)
  x = fmaxf(x, dppf<0x4E>(x));    // quad_perm [2,3,0,1]  (^2)
  x = fmaxf(x, dppf<0x141>(x));   // row_half_mirror      (^7)
  x = fmaxf(x, dppf<0x140>(x));   // row_mirror           (^15)
  return x;
}
__device__ __forceinline__ float red_sum16(float x) {
  x += dppf<0xB1>(x);
  x += dppf<0x4E>(x);
  x += dppf<0x141>(x);
  x += dppf<0x140>(x);
  return x;
}
#else
__device__ __forceinline__ float red_max16(float x) {
#pragma unroll
  for (int off = 1; off < 16; off <<= 1) x = fmaxf(x, __shfl_xor(x, off, 32));
  return x;
}
__device__ __forceinline__ float red_sum16(float x) {
#pragma unroll
  for (int off = 1; off < 16; off <<= 1) x += __shfl_xor(x, off, 32);
  return x;
}
#endif

// window row (win*256+tok) -> original row (b*256 + h*16 + w)
__device__ __forceinline__ int win_src_row(int rw) {
  int win = rw >> 8, tok = rw & 255;
  int b0 = win >> 4, h0 = (win >> 2) & 3, w0 = win & 3;
  int wb = tok >> 4, wh = (tok >> 2) & 3, ww = tok & 3;
  int b = b0 * 16 + wb, h = h0 * 4 + wh, w = w0 * 4 + ww;
  return (b << 8) + (h << 4) + w;
}

// ---------------------------------------------- f32 -> bf16, transposed
// out[n*K + k] = (bf16) in[k*N + n] ; 32x32 tiles through padded LDS
__global__ __launch_bounds__(256)
void cvt_transpose_kernel(const float* __restrict__ in, bf16* __restrict__ out,
                          int K, int N) {
  __shared__ float tile[32][33];
  const int bk = blockIdx.x * 32;         // k base
  const int bn = blockIdx.y * 32;         // n base
  const int tx = threadIdx.x & 31, ty = threadIdx.x >> 5;   // 32 x 8
#pragma unroll
  for (int i = 0; i < 32; i += 8)
    tile[ty + i][tx] = in[(size_t)(bk + ty + i) * N + bn + tx];
  __syncthreads();
#pragma unroll
  for (int i = 0; i < 32; i += 8)
    out[(size_t)(bn + ty + i) * K + bk + tx] = (bf16)tile[tx][ty + i];
}

// ------------------------------------------------- LayerNorm (rows of 768)
template<int GATHER>
__global__ __launch_bounds__(256)
void ln_kernel(const float* __restrict__ X, const float* __restrict__ g,
               const float* __restrict__ b, bf16* __restrict__ O) {
  int row = blockIdx.x;
  int src = GATHER ? win_src_row(row) : row;
  const float* xr = X + (size_t)src * 768;
  int t = threadIdx.x;
  float vals[3];
  float s = 0.f, q = 0.f;
#pragma unroll
  for (int i = 0; i < 3; ++i) {
    float v = xr[t + 256 * i];
    vals[i] = v; s += v; q += v * v;
  }
  __shared__ float rs[256], rq[256];
  rs[t] = s; rq[t] = q;
  __syncthreads();
  for (int off = 128; off > 0; off >>= 1) {
    if (t < off) { rs[t] += rs[t + off]; rq[t] += rq[t + off]; }
    __syncthreads();
  }
  float mean = rs[0] * (1.0f / 768.0f);
  float var  = rq[0] * (1.0f / 768.0f) - mean * mean;
  float rinv = rsqrtf(var + 1e-5f);
  bf16* orow = O + (size_t)row * 768;
#pragma unroll
  for (int i = 0; i < 3; ++i) {
    int c = t + 256 * i;
    orow[c] = (bf16)((vals[i] - mean) * rinv * g[c] + b[c]);
  }
}

// ------------------------------------------------------ bf16 WMMA GEMM
// C[M,N] = A[M,K] @ Bt[N,K]^T  (B pre-transposed). Both tiles staged by
// async global->LDS DMA, double-buffered, pipelined with s_wait_asynccnt.
enum { EPI_BF16_BIAS = 0, EPI_F32_RESID = 1, EPI_F32_RESID_SCATTER = 2,
       EPI_BF16_GELU = 3 };

template<int EPI>
__global__ __launch_bounds__(256)
void gemm_bf16_kernel(const bf16* __restrict__ A, const bf16* __restrict__ Bt,
                      const float* __restrict__ bias, void* __restrict__ Cout,
                      const float* __restrict__ R, int M, int N, int K) {
  constexpr int BM = 128, BN = 128, BK = 32, LDSS = 40;  // padded stride
  __shared__ bf16 As[2][BM][LDSS];   // async DMA targets, double-buffered
  __shared__ bf16 Bs[2][BN][LDSS];   // Bs[n][k] (Bt is already [N][K])
  const int tid = threadIdx.x;
  const int lane = tid & 31, wid = tid >> 5;
  const int l16 = lane & 15, half = lane >> 4;
  const int wm = wid & 3, wn = wid >> 2;
  const int bm0 = blockIdx.y * BM, bn0 = blockIdx.x * BN;

  v8f acc[2][4];
#pragma unroll
  for (int i = 0; i < 2; ++i)
#pragma unroll
    for (int j = 0; j < 4; ++j) acc[i][j] = v8f_zero();

  const int arow = tid >> 1, aseg = tid & 1;   // 128 rows x 32 k, 32B/thread

  auto issue_tile = [&](int kt, int pbuf) {
    const bf16* ga = A + (size_t)(bm0 + arow) * K + kt + aseg * 16;
    bf16* la = &As[pbuf][arow][aseg * 16];
    async_copy16(ga, la);
    async_copy16(ga + 8, la + 8);
    const bf16* gb = Bt + (size_t)(bn0 + arow) * K + kt + aseg * 16;
    bf16* lb = &Bs[pbuf][arow][aseg * 16];
    async_copy16(gb, lb);
    async_copy16(gb + 8, lb + 8);
  };

  issue_tile(0, 0);   // prologue

  int pb = 0;
  for (int kt = 0; kt < K; kt += BK, pb ^= 1) {
    if (kt + BK < K) {
      issue_tile(kt + BK, pb ^ 1);  // keep next tile's DMA in flight
      wait_async_le<4>();           // tile kt done; tile kt+BK still flying
    } else {
      wait_async_le<0>();
    }
    __syncthreads();

    v16bf af[2], bfr[4];
    const int khalf = half * 8;    // A frag: k = khalf..+7, khalf+16..+23
    const int kbase = half * 8;    // B frag: k = kbase..+7, kbase+16..+23
#pragma unroll
    for (int sm = 0; sm < 2; ++sm) {
      int r = wm * 32 + sm * 16 + l16;
      union { v16bf v; uint4 u[2]; } t;
      t.u[0] = *(const uint4*)&As[pb][r][khalf];
      t.u[1] = *(const uint4*)&As[pb][r][khalf + 16];
      af[sm] = t.v;
    }
#pragma unroll
    for (int sn = 0; sn < 4; ++sn) {
      int c = wn * 64 + sn * 16 + l16;
      union { v16bf v; uint4 u[2]; } t;
      t.u[0] = *(const uint4*)&Bs[pb][c][kbase];
      t.u[1] = *(const uint4*)&Bs[pb][c][kbase + 16];
      bfr[sn] = t.v;
    }
#pragma unroll
    for (int sm = 0; sm < 2; ++sm)
#pragma unroll
      for (int sn = 0; sn < 4; ++sn)
        acc[sm][sn] = wmma_bf16(af[sm], bfr[sn], acc[sm][sn]);
    __syncthreads();
  }

  // epilogue: C layout lane -> (n = l16, m = r + 8*half)
#pragma unroll
  for (int sm = 0; sm < 2; ++sm) {
#pragma unroll
    for (int sn = 0; sn < 4; ++sn) {
      int gc = bn0 + wn * 64 + sn * 16 + l16;
      float bv = bias[gc];
#pragma unroll
      for (int r = 0; r < 8; ++r) {
        int gr = bm0 + wm * 32 + sm * 16 + r + 8 * half;
        float v = acc[sm][sn][r] + bv;
        if (EPI == EPI_BF16_BIAS) {
          ((bf16*)Cout)[(size_t)gr * N + gc] = (bf16)v;
        } else if (EPI == EPI_BF16_GELU) {
          float gu = 0.5f * v * (1.0f + erff(v * 0.70710678118f));
          ((bf16*)Cout)[(size_t)gr * N + gc] = (bf16)gu;
        } else if (EPI == EPI_F32_RESID) {
          ((float*)Cout)[(size_t)gr * N + gc] = R[(size_t)gr * N + gc] + v;
        } else {  // window reverse fused into store
          int dr = win_src_row(gr);
          ((float*)Cout)[(size_t)dr * N + gc] = R[(size_t)dr * N + gc] + v;
        }
      }
    }
  }
}

// ------------------------------------------- flash attention, WMMA, hd=64
__global__ __launch_bounds__(256)
void attn_kernel(const bf16* __restrict__ Q, const bf16* __restrict__ Km,
                 const bf16* __restrict__ V, bf16* __restrict__ Y) {
  constexpr int T = 256, D = 768, HD = 64;
  const int rowbase = blockIdx.x * T;
  const int hc = blockIdx.y * HD;
  __shared__ bf16 Ks[T][HD + 8];    // [token][k], row = 128B data + 16B pad
  __shared__ bf16 Vt[HD][T + 8];    // [d][token]
  __shared__ bf16 Ps[8][16][40];    // per-wave P re-layout scratch
  const int tid = threadIdx.x, lane = tid & 31, wid = tid >> 5;
  const int l16 = lane & 15, half = lane >> 4;

#if HAS_TDM
  if (wid == 0) {  // uniform branch: one wave drives the TDM
    unsigned long long ga =
        (unsigned long long)(uintptr_t)(Km + (size_t)(rowbase)*D + hc);
    unsigned int lds = (unsigned int)(uintptr_t)(void*)&Ks[0][0];
    u32x4 g0;
    g0[0] = 1u;                                   // count=1 (valid D#)
    g0[1] = lds;                                  // lds_addr
    g0[2] = (unsigned int)ga;                     // global_addr[31:0]
    g0[3] = (unsigned int)((ga >> 32) & 0x1FFFFFFu) | (2u << 30);  // +type=2
    i32x8 g1;
    // data_size=2B, pad_enable, pad_interval: 32 dwords (code 4),
    // pad_amount: 4 dwords (code 3)
    g1[0] = (int)((1u << 16) | (1u << 20) | (4u << 22) | (3u << 25));
    g1[1] = (int)(768u << 16);       // tensor_dim0 = 768 (row length)
    g1[2] = (int)(0x8000u << 16);    // tensor_dim1 = 32768 rows
    g1[3] = (int)(64u << 16);        // tile_dim0 = 64 elements
    g1[4] = 256;                     // tile_dim1 = 256 rows
    g1[5] = 768;                     // tensor_dim0_stride = 768
    g1[6] = 0; g1[7] = 0;
    i32x4 gz = {0, 0, 0, 0};
#if __clang_major__ >= 23
    i32x8 gz8 = {0, 0, 0, 0, 0, 0, 0, 0};
    __builtin_amdgcn_tensor_load_to_lds(g0, g1, gz, gz, gz8, 0);
#else
    __builtin_amdgcn_tensor_load_to_lds(g0, g1, gz, gz, 0);
#endif
    __builtin_amdgcn_s_wait_tensorcnt(0);
  }
#else
  {
    const uint4* gk = (const uint4*)(Km + (size_t)(rowbase + tid) * D + hc);
#pragma unroll
    for (int i = 0; i < 8; ++i) *(uint4*)&Ks[tid][i * 8] = gk[i];
  }
#endif
  {  // V^T staging (transpose -> not TDM-able)
    union { uint4 u[8]; bf16 e[64]; } tv;
    const uint4* gv = (const uint4*)(V + (size_t)(rowbase + tid) * D + hc);
#pragma unroll
    for (int i = 0; i < 8; ++i) tv.u[i] = gv[i];
#pragma unroll
    for (int d = 0; d < 64; ++d) Vt[d][tid] = tv.e[d];
  }
  __syncthreads();

  const float scale = 0.125f;  // 1/sqrt(64)
  const int khalf = half * 8;

  for (int qt = wid; qt < 16; qt += 8) {
    const int m0 = qt * 16;
    v16bf qf[2];  // Q A-fragments kept in registers
#pragma unroll
    for (int kf = 0; kf < 2; ++kf) {
      const bf16* qp = Q + (size_t)(rowbase + m0 + l16) * D + hc + kf * 32;
      union { v16bf v; uint4 u[2]; } t;
      t.u[0] = *(const uint4*)(qp + khalf);
      t.u[1] = *(const uint4*)(qp + khalf + 16);
      qf[kf] = t.v;
    }
    float mrow[8], lrow[8];
#pragma unroll
    for (int r = 0; r < 8; ++r) { mrow[r] = -1e30f; lrow[r] = 0.f; }
    v8f oacc[4];
#pragma unroll
    for (int sn = 0; sn < 4; ++sn) oacc[sn] = v8f_zero();

    for (int j2 = 0; j2 < 8; ++j2) {  // 32 kv tokens per iteration
      v8f s[2];
#pragma unroll
      for (int jj = 0; jj < 2; ++jj) {
        const int tok = (j2 * 2 + jj) * 16 + l16;
        union { v16bf v; uint4 u[2]; } b0, b1;
        b0.u[0] = *(const uint4*)&Ks[tok][khalf];
        b0.u[1] = *(const uint4*)&Ks[tok][khalf + 16];
        b1.u[0] = *(const uint4*)&Ks[tok][32 + khalf];
        b1.u[1] = *(const uint4*)&Ks[tok][32 + khalf + 16];
        v8f z = v8f_zero();
        z = wmma_bf16(qf[0], b0.v, z);
        s[jj] = wmma_bf16(qf[1], b1.v, z);
      }
      // online softmax; all cross-lane reductions via DPP16 (pure VALU)
#pragma unroll
      for (int r = 0; r < 8; ++r) {
        float a  = s[0][r] * scale;
        float b2 = s[1][r] * scale;
        float mx = red_max16(fmaxf(a, b2));
        float mnew = fmaxf(mrow[r], mx);
        float al = __expf(mrow[r] - mnew);
        float p0 = __expf(a - mnew), p1 = __expf(b2 - mnew);
        mrow[r] = mnew;
        float ls = red_sum16(p0 + p1);
        lrow[r] = lrow[r] * al + ls;
        const int mrw = r + 8 * half;
        Ps[wid][mrw][l16]      = (bf16)p0;
        Ps[wid][mrw][16 + l16] = (bf16)p1;
#pragma unroll
        for (int sn = 0; sn < 4; ++sn) oacc[sn][r] *= al;
      }
      asm volatile("s_wait_dscnt 0" ::: "memory");  // cross-lane LDS visibility
      union { v16bf v; uint4 u[2]; } pt;            // reload P as A-fragment
      pt.u[0] = *(const uint4*)&Ps[wid][l16][khalf];
      pt.u[1] = *(const uint4*)&Ps[wid][l16][khalf + 16];
#pragma unroll
      for (int sn = 0; sn < 4; ++sn) {
        const bf16* vp = &Vt[sn * 16 + l16][j2 * 32 + khalf];
        union { v16bf v; uint4 u[2]; } bt;
        bt.u[0] = *(const uint4*)vp;
        bt.u[1] = *(const uint4*)(vp + 16);
        oacc[sn] = wmma_bf16(pt.v, bt.v, oacc[sn]);
      }
    }
#pragma unroll
    for (int sn = 0; sn < 4; ++sn) {
#pragma unroll
      for (int r = 0; r < 8; ++r) {
        float o = oacc[sn][r] / lrow[r];
        const int gr = rowbase + m0 + r + 8 * half;
        const int gc = hc + sn * 16 + l16;
        Y[(size_t)gr * D + gc] = (bf16)o;
      }
    }
  }
}

// ---------------------------------------------------------------- launcher
extern "C" void kernel_launch(void* const* d_in, const int* in_sizes, int n_in,
                              void* d_out, int out_size, void* d_ws, size_t ws_size,
                              hipStream_t stream) {
  (void)in_sizes; (void)n_in; (void)out_size; (void)ws_size;
  const float* x     = (const float*)d_in[0];
  const float* ln1_g = (const float*)d_in[2];  const float* ln1_b = (const float*)d_in[3];
  const float* ln2_g = (const float*)d_in[4];  const float* ln2_b = (const float*)d_in[5];
  const float* ln3_g = (const float*)d_in[6];  const float* ln3_b = (const float*)d_in[7];
  const float* q1_b = (const float*)d_in[9];
  const float* k1_b = (const float*)d_in[11];
  const float* v1_b = (const float*)d_in[13];
  const float* o1_b = (const float*)d_in[15];
  const float* q2_b = (const float*)d_in[17];
  const float* k2_b = (const float*)d_in[19];
  const float* v2_b = (const float*)d_in[21];
  const float* o2_b = (const float*)d_in[23];
  const float* fc1_b = (const float*)d_in[25];
  const float* fc2_b = (const float*)d_in[27];

  constexpr size_t DD = 768, ROWS = 32768;
  constexpr size_t WSQ = 768 * 768, WFC = 768 * 3072;

  bf16* wq1  = (bf16*)d_ws;        // all weights stored TRANSPOSED [N][K] bf16
  bf16* wk1  = wq1  + WSQ;
  bf16* wv1  = wk1  + WSQ;
  bf16* wo1  = wv1  + WSQ;
  bf16* wq2  = wo1  + WSQ;
  bf16* wk2  = wq2  + WSQ;
  bf16* wv2  = wk2  + WSQ;
  bf16* wo2  = wv2  + WSQ;
  bf16* wfc1 = wo2  + WSQ;
  bf16* wfc2 = wfc1 + WFC;
  bf16* Abuf = wfc2 + WFC;
  bf16* Qb   = Abuf + ROWS * DD;
  bf16* Kb   = Qb   + ROWS * DD;
  bf16* Vb   = Kb   + ROWS * DD;
  bf16* Yb   = Vb   + ROWS * DD;
  float* X1  = (float*)(Yb + ROWS * DD);
  bf16* Hb   = Qb;   // MLP hidden reuses Q/K/V/Y region

  dim3 blk(256);
  dim3 gN6(6, 256);
  dim3 gN24(24, 256);

  // weights -> bf16, transposed to [N][K]
  auto CVT_T = [&](int idx, bf16* dst, int K, int N) {
    cvt_transpose_kernel<<<dim3(K / 32, N / 32), 256, 0, stream>>>(
        (const float*)d_in[idx], dst, K, N);
  };
  CVT_T(8,  wq1,  768, 768);  CVT_T(10, wk1, 768, 768);
  CVT_T(12, wv1,  768, 768);  CVT_T(14, wo1, 768, 768);
  CVT_T(16, wq2,  768, 768);  CVT_T(18, wk2, 768, 768);
  CVT_T(20, wv2,  768, 768);  CVT_T(22, wo2, 768, 768);
  CVT_T(24, wfc1, 768, 3072); CVT_T(26, wfc2, 3072, 768);

  // ---- attn1 ----
  ln_kernel<0><<<32768, blk, 0, stream>>>(x, ln1_g, ln1_b, Abuf);
  gemm_bf16_kernel<EPI_BF16_BIAS><<<gN6, blk, 0, stream>>>(Abuf, wq1, q1_b, Qb, nullptr, 32768, 768, 768);
  gemm_bf16_kernel<EPI_BF16_BIAS><<<gN6, blk, 0, stream>>>(Abuf, wk1, k1_b, Kb, nullptr, 32768, 768, 768);
  gemm_bf16_kernel<EPI_BF16_BIAS><<<gN6, blk, 0, stream>>>(Abuf, wv1, v1_b, Vb, nullptr, 32768, 768, 768);
  attn_kernel<<<dim3(128, 12), blk, 0, stream>>>(Qb, Kb, Vb, Yb);
  gemm_bf16_kernel<EPI_F32_RESID><<<gN6, blk, 0, stream>>>(Yb, wo1, o1_b, (void*)X1, x, 32768, 768, 768);

  // ---- attn2 (window attention; gather fused into LN, scatter into o2) ----
  ln_kernel<1><<<32768, blk, 0, stream>>>(X1, ln2_g, ln2_b, Abuf);
  gemm_bf16_kernel<EPI_BF16_BIAS><<<gN6, blk, 0, stream>>>(Abuf, wq2, q2_b, Qb, nullptr, 32768, 768, 768);
  gemm_bf16_kernel<EPI_BF16_BIAS><<<gN6, blk, 0, stream>>>(Abuf, wk2, k2_b, Kb, nullptr, 32768, 768, 768);
  gemm_bf16_kernel<EPI_BF16_BIAS><<<gN6, blk, 0, stream>>>(Abuf, wv2, v2_b, Vb, nullptr, 32768, 768, 768);
  attn_kernel<<<dim3(128, 12), blk, 0, stream>>>(Qb, Kb, Vb, Yb);
  gemm_bf16_kernel<EPI_F32_RESID_SCATTER><<<gN6, blk, 0, stream>>>(Yb, wo2, o2_b, d_out, X1, 32768, 768, 768);

  // ---- MLP ----
  ln_kernel<0><<<32768, blk, 0, stream>>>((const float*)d_out, ln3_g, ln3_b, Abuf);
  gemm_bf16_kernel<EPI_BF16_GELU><<<gN24, blk, 0, stream>>>(Abuf, wfc1, fc1_b, Hb, nullptr, 32768, 3072, 768);
  gemm_bf16_kernel<EPI_F32_RESID><<<gN6, blk, 0, stream>>>(Hb, wfc2, fc2_b, d_out, (const float*)d_out, 32768, 768, 3072);
}
